// NetVLAD_34694745817828
// MI455X (gfx1250) — matrix operand order repaired and checked by reading, not compiled, not run on hardware
//
#include <hip/hip_runtime.h>
#include <math.h>

typedef __attribute__((ext_vector_type(16))) _Float16 v16h;
typedef __attribute__((ext_vector_type(8)))  _Float16 v8h;
typedef __attribute__((ext_vector_type(8)))  float    v8f;

#define Nimg 64
#define Dch  128
#define Kcl  64
#define Ptot 4800
#define PT   32        // pixels per chunk
#define NCHUNK (Ptot / PT)   // 150

// LDS strides (elements), padded to dodge bank conflicts; all row strides are
// multiples of 8 halfs (16B) so v8h loads stay 16B-aligned.
#define SW_STRIDE 136   // W[k][d]   f16 (128+8)
#define XN_STRIDE 40    // Xn[d][p]  f16 (32+8)
#define XT_STRIDE 136   // XnT[p][d] f16 (128+8)
#define SL_STRIDE 40    // logits[k][p] f32
#define SA_STRIDE 40    // sa[k][p]  f16

// ---- WMMA operand loaders (ISA 7.12.2 layouts) --------------------------
// A (16x32 f16): lane<16 -> row m=lane, halfs = K[c0..c0+7] ++ K[c0+16..c0+23]
//                lane>=16 -> row m=lane-16, halfs = K[c0+8..]++K[c0+24..]
__device__ __forceinline__ v16h load_a16(const _Float16* base, int stride,
                                         int m0, int c0, int lane) {
  int m   = lane & 15;
  int klo = (lane < 16) ? 0 : 8;
  const _Float16* p0 = base + (m0 + m) * stride + c0 + klo;
  v8h lo = *(const v8h*)(p0);
  v8h hi = *(const v8h*)(p0 + 16);
  v16h r;
#pragma unroll
  for (int i = 0; i < 8; ++i) { r[i] = lo[i]; r[8 + i] = hi[i]; }
  return r;
}

// B (32x16 f16) from transposed storage T[col][k]:
// lane<16 -> col n=lane, halfs = K[c0..c0+7] ++ K[c0+16..c0+23]; lane>=16 -> col n=lane-16, K[c0+8..]++K[c0+24..]
__device__ __forceinline__ v16h load_b16(const _Float16* baseT, int stride,
                                         int c0, int n0, int lane) {
  int n    = lane & 15;
  int koff = (lane < 16) ? 0 : 8;
  const _Float16* p0 = baseT + (n0 + n) * stride + c0 + koff;
  v8h lo = *(const v8h*)(p0);
  v8h hi = *(const v8h*)(p0 + 16);
  v16h r;
#pragma unroll
  for (int i = 0; i < 8; ++i) { r[i] = lo[i]; r[8 + i] = hi[i]; }
  return r;
}

// ---- Kernel 1: fused normalize + logits + softmax + aggregation ---------
__global__ __launch_bounds__(256)
void netvlad_fused_k1(const float* __restrict__ x, const float* __restrict__ w,
                      float* __restrict__ pAgg, float* __restrict__ pSum,
                      int S, int chunks_per_wg) {
  __shared__ __align__(16) _Float16 sW [Kcl * SW_STRIDE];
  __shared__ __align__(16) _Float16 sXn[Dch * XN_STRIDE];
  __shared__ __align__(16) _Float16 sXT[PT  * XT_STRIDE];
  __shared__ __align__(16) float    sL [Kcl * SL_STRIDE];
  __shared__ __align__(16) _Float16 sSA[Kcl * SA_STRIDE];
  __shared__ float sSS[8 * PT];    // 8-group partials (norm, softmax max/den)
  __shared__ float sInvN[PT];
  __shared__ float sMax[PT];
  __shared__ float sDen[PT];
  __shared__ float sSp[256];       // sa row-sum partials
  __shared__ float sSum[Kcl];

  const int tid  = threadIdx.x;
  const int lane = tid & 31;
  const int wave = tid >> 5;          // 0..7
  const int n    = blockIdx.x / S;    // image
  const int s    = blockIdx.x % S;    // slice

  // Load conv_w (64x128 f32) -> f16 LDS, once.
#pragma unroll
  for (int i = tid; i < Kcl * Dch; i += 256) {
    int k = i >> 7, d = i & 127;
    sW[k * SW_STRIDE + d] = (_Float16)w[i];
  }
  if (tid < Kcl) sSum[tid] = 0.0f;

  // Persistent aggregation accumulators: wave owns d-tile = wave, kt = 0..3.
  v8f acc[4];
#pragma unroll
  for (int kt = 0; kt < 4; ++kt)
#pragma unroll
    for (int j = 0; j < 8; ++j) acc[kt][j] = 0.0f;

  __syncthreads();

  const int p = tid & 31;   // pixel in chunk (phases 1 & 3)
  const int q = tid >> 5;   // 8-way group id
  const float* xbase = x + (size_t)n * Dch * Ptot;

  for (int j = 0; j < chunks_per_wg; ++j) {
    const int chunk = s + j * S;
    const int p0g   = chunk * PT;

    // ---- Phase 1: load chunk + per-pixel L2 norm ----
    float xv[16];
    float ss = 0.0f;
    const float* xb = xbase + p0g + p;
#pragma unroll
    for (int i = 0; i < 16; ++i) {
      float v = xb[(size_t)(q * 16 + i) * Ptot];
      xv[i] = v;
      ss += v * v;
    }
    if (j + 1 < chunks_per_wg) {  // global_prefetch_b8 for next chunk
      __builtin_prefetch(xbase + (size_t)(q * 16) * Ptot + (chunk + S) * PT + p, 0, 1);
    }
    sSS[q * PT + p] = ss;
    __syncthreads();
    if (tid < PT) {
      float t = 0.0f;
#pragma unroll
      for (int i = 0; i < 8; ++i) t += sSS[i * PT + tid];
      sInvN[tid] = 1.0f / fmaxf(sqrtf(t), 1e-12f);
    }
    __syncthreads();
    {
      float inv = sInvN[p];
      _Float16 hv[16];
#pragma unroll
      for (int i = 0; i < 16; ++i) hv[i] = (_Float16)(xv[i] * inv);
      // contiguous 16 halfs of XnT row -> two ds_store_b128
      v8h h0, h1;
#pragma unroll
      for (int i = 0; i < 8; ++i) { h0[i] = hv[i]; h1[i] = hv[8 + i]; }
      *(v8h*)&sXT[p * XT_STRIDE + q * 16]     = h0;
      *(v8h*)&sXT[p * XT_STRIDE + q * 16 + 8] = h1;
      // transpose side: scattered b16 stores
#pragma unroll
      for (int i = 0; i < 16; ++i)
        sXn[(q * 16 + i) * XN_STRIDE + p] = hv[i];
    }
    __syncthreads();

    // ---- Phase 2: logits = W(64x128) * Xn(128x32), f16 WMMA ----
    {
      int kt = wave >> 1;   // 0..3
      int pt = wave & 1;    // 0..1
      v8f c = {};
#pragma unroll
      for (int d0 = 0; d0 < Dch; d0 += 32) {
        v16h a = load_a16(sW,  SW_STRIDE, kt * 16, d0, lane);
        v16h b = load_b16(sXT, XT_STRIDE, d0, pt * 16, lane);
        c = __builtin_amdgcn_wmma_f32_16x16x32_f16(false, a, false, b,
                                                   (short)0, c, false, false);
      }
      int col   = lane & 15;
      int rbase = (lane < 16) ? 0 : 8;
#pragma unroll
      for (int r = 0; r < 8; ++r)
        sL[(kt * 16 + rbase + r) * SL_STRIDE + pt * 16 + col] = c[r];
    }
    __syncthreads();

    // ---- Phase 3: softmax over K per pixel, parallel over all 256 threads.
    // Thread (p = tid&31, q = tid>>5) owns clusters k in [8q, 8q+8).
    {
      float m = -3.4e38f;
#pragma unroll
      for (int i = 0; i < 8; ++i)
        m = fmaxf(m, sL[(q * 8 + i) * SL_STRIDE + p]);
      sSS[q * PT + p] = m;
    }
    __syncthreads();
    if (tid < PT) {
      float m = sSS[tid];
#pragma unroll
      for (int i = 1; i < 8; ++i) m = fmaxf(m, sSS[i * PT + tid]);
      sMax[tid] = m;
    }
    __syncthreads();
    {
      float m = sMax[p];
      float den = 0.0f;
#pragma unroll
      for (int i = 0; i < 8; ++i) {
        int idx = (q * 8 + i) * SL_STRIDE + p;
        float e = __expf(sL[idx] - m);
        sL[idx] = e;
        den += e;
      }
      sSS[q * PT + p] = den;
    }
    __syncthreads();
    if (tid < PT) {
      float t = 0.0f;
#pragma unroll
      for (int i = 0; i < 8; ++i) t += sSS[i * PT + tid];
      sDen[tid] = 1.0f / t;
    }
    __syncthreads();
    {
      float r = sDen[p];
#pragma unroll
      for (int i = 0; i < 8; ++i)
        sSA[(q * 8 + i) * SA_STRIDE + p] = (_Float16)(sL[(q * 8 + i) * SL_STRIDE + p] * r);
    }
    __syncthreads();

    // sa row-sum partials: thread (k = tid>>2, pg = tid&3) sums 8 pixels.
    {
      int k  = tid >> 2;
      int pg = tid & 3;
      float t = 0.0f;
#pragma unroll
      for (int i = 0; i < 8; ++i) t += (float)sSA[k * SA_STRIDE + pg * 8 + i];
      sSp[tid] = t;
    }

    // ---- Phase 4: agg[k][d] += SA(64x32) * Xn^T(32x128), f16 WMMA ----
    {
      int dt = wave;  // 0..7
      v16h b = load_b16(sXn, XN_STRIDE, 0, dt * 16, lane);
#pragma unroll
      for (int kt = 0; kt < 4; ++kt) {
        v16h a = load_a16(sSA, SA_STRIDE, kt * 16, 0, lane);
        acc[kt] = __builtin_amdgcn_wmma_f32_16x16x32_f16(false, a, false, b,
                                                         (short)0, acc[kt],
                                                         false, false);
      }
    }
    __syncthreads();
    if (tid < Kcl)
      sSum[tid] += sSp[tid * 4] + sSp[tid * 4 + 1] + sSp[tid * 4 + 2] + sSp[tid * 4 + 3];
    __syncthreads();   // protect LDS before next chunk overwrites it
  }

  // ---- Write partials ----
  {
    float* aggOut = pAgg + (size_t)(n * S + s) * Kcl * Dch;
    int col   = lane & 15;
    int rbase = (lane < 16) ? 0 : 8;
#pragma unroll
    for (int kt = 0; kt < 4; ++kt)
#pragma unroll
      for (int r = 0; r < 8; ++r)
        aggOut[(kt * 16 + rbase + r) * Dch + wave * 16 + col] = acc[kt][r];
  }
  if (tid < Kcl) pSum[(size_t)(n * S + s) * Kcl + tid] = sSum[tid];
}

// ---- Kernel 2: reduce partials, subtract centroids, double L2 norm ------
__global__ __launch_bounds__(256)
void netvlad_norm_k2(const float* __restrict__ pAgg, const float* __restrict__ pSum,
                     const float* __restrict__ cent, float* __restrict__ out, int S) {
  __shared__ float sKp[Kcl * 4];
  __shared__ float sInvK[Kcl];
  __shared__ float sRed[256];
  __shared__ float sTot;

  const int n   = blockIdx.x;
  const int tid = threadIdx.x;
  const int k   = tid >> 2;
  const int g   = tid & 3;
  const int d0  = g * 32;

  float ssum = 0.0f;
  for (int s = 0; s < S; ++s) ssum += pSum[(size_t)(n * S + s) * Kcl + k];

  float v[32];
#pragma unroll
  for (int jj = 0; jj < 32; ++jj) v[jj] = -ssum * cent[k * Dch + d0 + jj];
  for (int s = 0; s < S; ++s) {
    const float* a = pAgg + ((size_t)(n * S + s) * Kcl + k) * Dch + d0;
#pragma unroll
    for (int jj = 0; jj < 32; ++jj) v[jj] += a[jj];
  }

  float ss = 0.0f;
#pragma unroll
  for (int jj = 0; jj < 32; ++jj) ss += v[jj] * v[jj];
  sKp[k * 4 + g] = ss;
  __syncthreads();
  if (tid < Kcl) {
    float t = sKp[tid * 4] + sKp[tid * 4 + 1] + sKp[tid * 4 + 2] + sKp[tid * 4 + 3];
    sInvK[tid] = 1.0f / fmaxf(sqrtf(t), 1e-12f);
  }
  __syncthreads();
  float ik  = sInvK[k];
  float gsq = 0.0f;
#pragma unroll
  for (int jj = 0; jj < 32; ++jj) { v[jj] *= ik; gsq += v[jj] * v[jj]; }
  sRed[tid] = gsq;
  __syncthreads();
  if (tid == 0) {
    float t = 0.0f;
    for (int i = 0; i < 256; ++i) t += sRed[i];
    sTot = t;
  }
  __syncthreads();
  float ig = 1.0f / fmaxf(sqrtf(sTot), 1e-12f);
  float* o = out + (size_t)n * Kcl * Dch + k * Dch + d0;
#pragma unroll
  for (int jj = 0; jj < 32; ++jj) o[jj] = v[jj] * ig;
}

extern "C" void kernel_launch(void* const* d_in, const int* in_sizes, int n_in,
                              void* d_out, int out_size, void* d_ws, size_t ws_size,
                              hipStream_t stream) {
  const float* x    = (const float*)d_in[0];   // [64,128,60,80]
  const float* w    = (const float*)d_in[1];   // [64,128]
  const float* cent = (const float*)d_in[2];   // [64,128]
  float* out        = (float*)d_out;           // [64, 64*128]

  // Pick the largest slice count that fits the workspace (divisors of 150).
  int S = 1;
  const int cands[4] = {15, 5, 3, 1};
  for (int i = 0; i < 4; ++i) {
    size_t need = (size_t)Nimg * cands[i] * (Kcl * Dch + Kcl) * sizeof(float);
    if (need <= ws_size) { S = cands[i]; break; }
  }
  const int chunks_per_wg = NCHUNK / S;

  float* pAgg = (float*)d_ws;
  float* pSum = pAgg + (size_t)Nimg * S * Kcl * Dch;

  netvlad_fused_k1<<<Nimg * S, 256, 0, stream>>>(x, w, pAgg, pSum, S, chunks_per_wg);
  netvlad_norm_k2<<<Nimg, 256, 0, stream>>>(pAgg, pSum, cent, out, S);
}
